// ResBlock_2327872274545
// MI455X (gfx1250) — matrix operand order repaired and checked by reading, compiled, not run on hardware
//
#include <hip/hip_runtime.h>
#include <hip/hip_bf16.h>
#include <math.h>

#define N_NODES 10000
#define N_EDGES 320000
#define HDIM    256
#define EDIM    64
#define NT_ALL  16   // total 16-col tiles in HDIM
#define NTW     4    // col tiles per wave in GEMM
#define NGRP    4    // col groups (NT_ALL / NTW)

typedef __attribute__((ext_vector_type(16))) __bf16 v16bf;
typedef __attribute__((ext_vector_type(8)))  float  v8f;

// gelu_tanh(x) = 0.5x(1+tanh(u)) = x * sigmoid(2u), u = 0.79788456(x + 0.044715x^3)
// Branch-free: v_exp_f32 + v_rcp_f32.
__device__ __forceinline__ float gelu_f(float x) {
    float t = 1.5957691216057308f * (x + 0.044715f * x * x * x);   // 2u
    float e = __builtin_amdgcn_exp2f(-1.4426950408889634f * t);    // exp(-2u)
    return x * __builtin_amdgcn_rcpf(1.0f + e);
}

// ---------------------------------------------------------------------------
// A-operand (16x32 bf16, MxK), contiguous K order (matches swizzled B):
//   lane L (L<16): row M=L,   elems 0-15 -> K=k0+0..15
//   lane L (>=16): row M=L-16,elems 0-15 -> K=k0+16..31
// arow points at this lane's f32 row (row M = lane&15), contiguous in K.
// ---------------------------------------------------------------------------
__device__ __forceinline__ v16bf load_a_tile(const float* __restrict__ arow, int k0, int half) {
    const float* p = arow + k0 + 16 * half;
    float4 f0 = *(const float4*)(p);
    float4 f1 = *(const float4*)(p + 4);
    float4 f2 = *(const float4*)(p + 8);
    float4 f3 = *(const float4*)(p + 12);
    v16bf a;
    a[0]  = (__bf16)f0.x; a[1]  = (__bf16)f0.y; a[2]  = (__bf16)f0.z; a[3]  = (__bf16)f0.w;
    a[4]  = (__bf16)f1.x; a[5]  = (__bf16)f1.y; a[6]  = (__bf16)f1.z; a[7]  = (__bf16)f1.w;
    a[8]  = (__bf16)f2.x; a[9]  = (__bf16)f2.y; a[10] = (__bf16)f2.z; a[11] = (__bf16)f2.w;
    a[12] = (__bf16)f3.x; a[13] = (__bf16)f3.y; a[14] = (__bf16)f3.z; a[15] = (__bf16)f3.w;
    return a;
}

// ---------------------------------------------------------------------------
// Weight pre-swizzle: W (f32, row-major [K, ldw]) -> Wp (bf16) packed so that
// for tile t = kt*16 + nt, lane L's 16 B-operand values (col N = L&15,
// K = kt*32 + 16*(L>=16) + e) are 32 contiguous bytes at Wp + t*512 + L*16.
// One wave per 32x16 tile; grid sized exactly.
// ---------------------------------------------------------------------------
__global__ void swizzle_w(const float* __restrict__ W, int ldw, int Ksteps,
                          __bf16* __restrict__ Wp) {
    int wid  = (blockIdx.x * blockDim.x + threadIdx.x) >> 5;   // tile id
    int lane = threadIdx.x & 31;
    int half = lane >> 4;
    int nlo  = lane & 15;
    int kt = wid >> 4;
    int nt = wid & 15;
    if (kt >= Ksteps) return;                                   // wave-uniform
    const float* p = W + (size_t)(kt * 32 + 16 * half) * ldw + nt * 16 + nlo;
    v16bf b;
#pragma unroll
    for (int e = 0; e < 16; ++e) b[e] = (__bf16)p[(size_t)e * ldw];
    *(v16bf*)(Wp + (size_t)wid * 512 + lane * 32 / 2) = b;      // lane*16 bf16
}

// ---------------------------------------------------------------------------
// GEMM: out[M,256] = act(A[M,K] @ W + bias) (+ resid), W pre-swizzled bf16.
// One wave per 16-row x 64-col strip (4 col tiles): per K-step, 4 A vector
// loads feed 4 WMMAs; each B operand is two global_load_b128. 2500 waves.
// C/D layout: lane L -> col N=L&15; VGPR v -> row M = v + 8*(L>=16).
// ---------------------------------------------------------------------------
template <bool GELU, bool RES>
__global__ void wmma_gemm(const float* __restrict__ A, int lda,
                          const __bf16* __restrict__ Wp,
                          const float* __restrict__ bias,
                          const float* __restrict__ resid,
                          float* __restrict__ out, int ldo, int Ksteps) {
    int wid  = (blockIdx.x * blockDim.x + threadIdx.x) >> 5;
    int lane = threadIdx.x & 31;
    int half = lane >> 4;
    int nlo  = lane & 15;
    int mt = wid / NGRP;           // row strip
    int ng = wid % NGRP;           // col group (4 tiles each)

    const float* arow = A + (size_t)(mt * 16 + nlo) * lda;
    const __bf16* wlane = Wp + (size_t)lane * 16 + (size_t)ng * NTW * 512;

    v8f acc[NTW];
#pragma unroll
    for (int j = 0; j < NTW; ++j) acc[j] = (v8f){};

    for (int ks = 0; ks < Ksteps; ++ks) {
        v16bf a = load_a_tile(arow, ks * 32, half);
        const __bf16* bp = wlane + (size_t)ks * 16 * 512;
#pragma unroll
        for (int j = 0; j < NTW; ++j) {
            v16bf b = *(const v16bf*)(bp + (size_t)j * 512);
            acc[j] = __builtin_amdgcn_wmma_f32_16x16x32_bf16(false, a, false, b,
                                                             (short)0, acc[j], false, false);
        }
    }

#pragma unroll
    for (int j = 0; j < NTW; ++j) {
        int ncol = (ng * NTW + j) * 16 + nlo;
        float bv = bias ? bias[ncol] : 0.0f;
#pragma unroll
        for (int v = 0; v < 8; ++v) {
            size_t row = (size_t)(mt * 16 + v + 8 * half);
            float o = acc[j][v] + bv;
            if (GELU) o = gelu_f(o);
            if (RES)  o += resid[row * ldo + ncol];
            out[row * ldo + ncol] = o;
        }
    }
}

// ---------------------------------------------------------------------------
// Edge message kernel. One wave per 16-edge tile:
//   E-tile = ef[16x64] @ We[64x256] via 2 WMMA per 16-col tile (pre-swizzled
//   bf16 We -> b128 operand loads), then
//   m = gelu(Anode[src] + Bnode[dst] + E + bias), f32 atomic scatter into agg.
// D element v of lane L corresponds to edge 8*half+v, col nt*16 + (L&15).
// ---------------------------------------------------------------------------
__global__ void edge_msg(const float* __restrict__ Anode,
                         const float* __restrict__ Bnode,
                         const float* __restrict__ ef,
                         const int* __restrict__ eidx,     // [2, N_EDGES] int32
                         const __bf16* __restrict__ Wep,   // swizzled [64,256]: 32 tiles
                         const float* __restrict__ bias,   // [256]
                         float* __restrict__ agg) {
    int wid  = (blockIdx.x * blockDim.x + threadIdx.x) >> 5;
    int lane = threadIdx.x & 31;
    int half = lane >> 4;
    int nlo  = lane & 15;

    size_t ebase = (size_t)wid * 16;

    // ef tile A-operands (row = edge ebase + (lane&15), K = 0..63)
    const float* arow = ef + (ebase + nlo) * EDIM;
    v16bf a0 = load_a_tile(arow, 0, half);
    v16bf a1 = load_a_tile(arow, 32, half);

    // Each lane owns output rows (edges) 8*half + 0..7 -> preload indices
    int srcs[8], dsts[8];
#pragma unroll
    for (int v = 0; v < 8; ++v) {
        size_t e = ebase + 8 * half + v;
        srcs[v] = eidx[e];
        dsts[v] = eidx[(size_t)N_EDGES + e];
    }

    const __bf16* wlane = Wep + (size_t)lane * 16;
    for (int nt = 0; nt < NT_ALL; ++nt) {
        v16bf b0 = *(const v16bf*)(wlane + (size_t)nt * 512);
        v16bf b1 = *(const v16bf*)(wlane + (size_t)(16 + nt) * 512);
        v8f acc = {};
        acc = __builtin_amdgcn_wmma_f32_16x16x32_bf16(false, a0, false, b0,
                                                      (short)0, acc, false, false);
        acc = __builtin_amdgcn_wmma_f32_16x16x32_bf16(false, a1, false, b1,
                                                      (short)0, acc, false, false);
        int ncol = nt * 16 + nlo;
        float bv = bias[ncol];
#pragma unroll
        for (int v = 0; v < 8; ++v) {
            float m = acc[v]
                    + Anode[(size_t)srcs[v] * HDIM + ncol]
                    + Bnode[(size_t)dsts[v] * HDIM + ncol]
                    + bv;
            m = gelu_f(m);
            unsafeAtomicAdd(&agg[(size_t)dsts[v] * HDIM + ncol], m);
        }
    }
}

// ---------------------------------------------------------------------------
extern "C" void kernel_launch(void* const* d_in, const int* in_sizes, int n_in,
                              void* d_out, int out_size, void* d_ws, size_t ws_size,
                              hipStream_t stream) {
    const float* x    = (const float*)d_in[0];
    const int*   ei   = (const int*)d_in[1];    // edge_index (int32 on device)
    const float* ef   = (const float*)d_in[2];
    const float* Wff1 = (const float*)d_in[3];
    const float* bff1 = (const float*)d_in[4];
    const float* Wmp1 = (const float*)d_in[5];  // [576,256] = [Wa;Wb;We]
    const float* bmp1 = (const float*)d_in[6];
    const float* Wmp2 = (const float*)d_in[7];
    const float* bmp2 = (const float*)d_in[8];
    const float* Wff2 = (const float*)d_in[9];
    const float* bff2 = (const float*)d_in[10];
    float* out = (float*)d_out;

    const size_t NH = (size_t)N_NODES * HDIM;
    float* h   = (float*)d_ws;   // node state
    float* Ab  = h + NH;         // h @ W_a
    float* Bb  = Ab + NH;        // h @ W_b
    float* agg = Bb + NH;        // residual-seeded aggregation

    // Pre-swizzled bf16 weights (tiles of 512 bf16 = 1 KB each)
    const size_t T256 = 128 * 512;              // K=256 slice: 8 ksteps * 16 col tiles
    const size_t T64  = 32 * 512;               // K=64  slice: 2 ksteps * 16 col tiles
    __bf16* wpFF1 = (__bf16*)(agg + NH);
    __bf16* wpA1  = wpFF1 + T256;
    __bf16* wpB1  = wpA1  + T256;
    __bf16* wpE1  = wpB1  + T256;
    __bf16* wpA2  = wpE1  + T64;
    __bf16* wpB2  = wpA2  + T256;
    __bf16* wpE2  = wpB2  + T256;
    __bf16* wpFF2 = wpE2  + T64;

    const dim3 blk(128);
    const size_t HH = (size_t)HDIM * HDIM;      // 65536 (floats per 256-row slice)

    // ---- weight pre-swizzle (runs once per launch; ~832 KB, L2-resident) ----
    swizzle_w<<<32, blk, 0, stream>>>(Wff1,            HDIM, 8, wpFF1);
    swizzle_w<<<32, blk, 0, stream>>>(Wmp1,            HDIM, 8, wpA1);
    swizzle_w<<<32, blk, 0, stream>>>(Wmp1 + HH,       HDIM, 8, wpB1);
    swizzle_w<<< 8, blk, 0, stream>>>(Wmp1 + 2 * HH,   HDIM, 2, wpE1);
    swizzle_w<<<32, blk, 0, stream>>>(Wmp2,            HDIM, 8, wpA2);
    swizzle_w<<<32, blk, 0, stream>>>(Wmp2 + HH,       HDIM, 8, wpB2);
    swizzle_w<<< 8, blk, 0, stream>>>(Wmp2 + 2 * HH,   HDIM, 2, wpE2);
    swizzle_w<<<32, blk, 0, stream>>>(Wff2,            HDIM, 8, wpFF2);

    const int gemm_blocks = (N_NODES / 16) * NGRP / 4;  // 625 blocks * 4 waves = 2500
    const int edge_blocks = (N_EDGES / 16) / 4;         // 5000 blocks * 4 waves

    // FFN1: h = gelu(x @ Wff1 + bff1)
    wmma_gemm<true, false><<<gemm_blocks, blk, 0, stream>>>(
        x, HDIM, wpFF1, bff1, nullptr, h, HDIM, 8);

    // ----- message passing layer 1 -----
    wmma_gemm<false, false><<<gemm_blocks, blk, 0, stream>>>(
        h, HDIM, wpA1, nullptr, nullptr, Ab, HDIM, 8);
    wmma_gemm<false, false><<<gemm_blocks, blk, 0, stream>>>(
        h, HDIM, wpB1, nullptr, nullptr, Bb, HDIM, 8);
    hipMemcpyAsync(agg, h, NH * sizeof(float), hipMemcpyDeviceToDevice, stream);
    edge_msg<<<edge_blocks, blk, 0, stream>>>(Ab, Bb, ef, ei, wpE1, bmp1, agg);
    // agg now holds h2 = h + segment_sum(messages)

    // ----- message passing layer 2 -----
    wmma_gemm<false, false><<<gemm_blocks, blk, 0, stream>>>(
        agg, HDIM, wpA2, nullptr, nullptr, Ab, HDIM, 8);
    wmma_gemm<false, false><<<gemm_blocks, blk, 0, stream>>>(
        agg, HDIM, wpB2, nullptr, nullptr, Bb, HDIM, 8);
    hipMemcpyAsync(h, agg, NH * sizeof(float), hipMemcpyDeviceToDevice, stream);
    edge_msg<<<edge_blocks, blk, 0, stream>>>(Ab, Bb, ef, ei, wpE2, bmp2, h);
    // h now holds h3

    // FFN2 + residual: out = x + h3 @ Wff2 + bff2
    wmma_gemm<false, true><<<gemm_blocks, blk, 0, stream>>>(
        h, HDIM, wpFF2, bff2, x, out, HDIM, 8);
}
